// FusedMoEModularKernel_25795573580291
// MI455X (gfx1250) — matrix operand order repaired and checked by reading, not compiled, not run on hardware
//
#include <hip/hip_runtime.h>

#define MTOK 4096
#define KDIM 2048
#define IDIM 2048
#define NEXP 8
#define NTOPK 2
#define MAXP (MTOK * NTOPK)      // worst-case rows per expert list
#define ROWS 32                  // routed rows per workgroup tile (2 row-groups)
#define MAXTILES (MAXP / ROWS)   // 256

typedef __attribute__((ext_vector_type(16))) __bf16 v16bf;
typedef __attribute__((ext_vector_type(8)))  float  v8f;

union FragU { uint4 q[2]; v16bf v; };

__device__ __forceinline__ unsigned short f32_to_bf16(float f) {
  unsigned int u = __float_as_uint(f);
  u += 0x7FFFu + ((u >> 16) & 1u);   // round-to-nearest-even
  return (unsigned short)(u >> 16);
}

__device__ __forceinline__ v8f v8f_zero() {
  v8f z = {0.f, 0.f, 0.f, 0.f, 0.f, 0.f, 0.f, 0.f};
  return z;
}

__device__ __forceinline__ v8f wmma_bf16(v16bf a, v16bf b, v8f c) {
  // (neg_a, A, neg_b, B, c_mod, C, reuse_a, reuse_b)
  return __builtin_amdgcn_wmma_f32_16x16x32_bf16(false, a, false, b,
                                                 (short)0, c, false, false);
}

// A-matrix 16x32 bf16 fragment (ISA 7.12.2): lane&15 = M row; lane-half adds
// +8 to K; VGPRs 0..3 hold K[0..7], VGPRs 4..7 hold K[16..23] -> 2x b128.
__device__ __forceinline__ v16bf load_a_frag(const unsigned short* rowbase,
                                             int k0, int hi) {
  FragU f;
  f.q[0] = *(const uint4*)(rowbase + k0 + hi * 8);
  f.q[1] = *(const uint4*)(rowbase + k0 + 16 + hi * 8);
  return f.v;
}

// B-matrix 32x16 bf16 fragment: lane&15 = N col; lane-half adds +8 to K;
// elements 0..7 hold K[0..7], elements 8..15 hold K[16..23] -> 2x b128.
__device__ __forceinline__ v16bf load_b_frag(const unsigned short* rowbase,
                                             int k0, int hi) {
  const unsigned short* p = rowbase + k0 + hi * 8;
  FragU f;
  f.q[0] = *(const uint4*)(p);
  f.q[1] = *(const uint4*)(p + 16);
  return f.v;
}

__global__ void zero_init(float* __restrict__ out, int* __restrict__ counts) {
  size_t i = (size_t)blockIdx.x * blockDim.x + threadIdx.x;
  ((float4*)out)[i] = make_float4(0.f, 0.f, 0.f, 0.f);
  if (blockIdx.x == 0 && threadIdx.x < NEXP) counts[threadIdx.x] = 0;
}

__global__ void cvt_f32_to_bf16(const float* __restrict__ src,
                                unsigned short* __restrict__ dst,
                                long long n) {
  long long i = ((long long)blockIdx.x * blockDim.x + threadIdx.x) * 4;
  if (i + 3 < n) {
    float4 f = *(const float4*)(src + i);
    unsigned int lo = (unsigned int)f32_to_bf16(f.x) |
                      ((unsigned int)f32_to_bf16(f.y) << 16);
    unsigned int hh = (unsigned int)f32_to_bf16(f.z) |
                      ((unsigned int)f32_to_bf16(f.w) << 16);
    *(uint2*)(dst + i) = make_uint2(lo, hh);
  }
}

__global__ void build_lists(const int* __restrict__ ids,
                            const float* __restrict__ tw,
                            int* __restrict__ counts,
                            int* __restrict__ tok,
                            float* __restrict__ wgt) {
  int idx = blockIdx.x * blockDim.x + threadIdx.x;   // 0 .. M*TOPK-1
  int e = ids[idx];
  int pos = atomicAdd(&counts[e], 1);
  tok[(size_t)e * MAXP + pos] = idx / NTOPK;
  wgt[(size_t)e * MAXP + pos] = tw[idx];
}

// Per workgroup: 32 routed rows of one expert. Full activation tile a
// (32 x 2048 bf16 = 128 KB) lives in LDS -- a CDNA5 320KB-WGP design point.
__global__ __launch_bounds__(256, 1) void moe_main(
    const unsigned short* __restrict__ Xbf,
    const unsigned short* __restrict__ W1bf,
    const unsigned short* __restrict__ W2bf,
    const int* __restrict__ counts,
    const int* __restrict__ tok,
    const float* __restrict__ wgt,
    float* __restrict__ out) {
  const int e = blockIdx.y;
  const int cnt = counts[e];
  if ((int)blockIdx.x * ROWS >= cnt) return;   // block-uniform

  const int tid = threadIdx.x;
  const int wave = tid >> 5;
  const int lane = tid & 31;
  const int mr = lane & 15;
  const int hi = lane >> 4;

  __shared__ __align__(16) unsigned short s_a[ROWS][IDIM];  // 128 KB bf16
  __shared__ int s_token[ROWS];
  __shared__ float s_w[ROWS];

  if (tid < ROWS) {
    int g = blockIdx.x * ROWS + tid;
    if (g < cnt) {
      s_token[tid] = tok[(size_t)e * MAXP + g];
      s_w[tid] = wgt[(size_t)e * MAXP + g];
    } else {             // padding row: token 0 with zero combine weight
      s_token[tid] = 0;
      s_w[tid] = 0.0f;
    }
  }
  __syncthreads();

  const unsigned short* w1e = W1bf + (size_t)e * (2 * IDIM) * KDIM;
  const unsigned short* w2e = W2bf + (size_t)e * KDIM * IDIM;
  const unsigned short* xRow0 = Xbf + (size_t)s_token[mr] * KDIM;
  const unsigned short* xRow1 = Xbf + (size_t)s_token[16 + mr] * KDIM;

  // ---- phase 1: a = silu(X@Wg^T) * (X@Wu^T), whole I dim into LDS ----
  // 128 gate col-tiles; each wave owns 16, processed as 8 pairs so every
  // A fragment feeds 4 WMMAs (2 col-tiles x gate/up).
  for (int p = 0; p < 8; ++p) {
    const int ct0 = (p * 8 + wave) * 2;          // col tiles ct0, ct0+1
    const unsigned short* bg[2];
    const unsigned short* bu[2];
    bg[0] = w1e + (size_t)(ct0 * 16 + mr) * KDIM;
    bg[1] = bg[0] + (size_t)16 * KDIM;
    bu[0] = w1e + (size_t)((IDIM + ct0 * 16) + mr) * KDIM;
    bu[1] = bu[0] + (size_t)16 * KDIM;
    __builtin_prefetch(bg[0], 0, 0);
    __builtin_prefetch(bg[1], 0, 0);
    __builtin_prefetch(bu[0], 0, 0);
    __builtin_prefetch(bu[1], 0, 0);

    v8f gA[2][2], uA[2][2];                      // [col-tile][row-group]
#pragma unroll
    for (int t = 0; t < 2; ++t)
#pragma unroll
      for (int rg = 0; rg < 2; ++rg) {
        gA[t][rg] = v8f_zero();
        uA[t][rg] = v8f_zero();
      }

    for (int k0 = 0; k0 < KDIM; k0 += 32) {
      v16bf a0 = load_a_frag(xRow0, k0, hi);
      v16bf a1 = load_a_frag(xRow1, k0, hi);
#pragma unroll
      for (int t = 0; t < 2; ++t) {
        v16bf bG = load_b_frag(bg[t], k0, hi);
        v16bf bU = load_b_frag(bu[t], k0, hi);
        gA[t][0] = wmma_bf16(a0, bG, gA[t][0]);
        uA[t][0] = wmma_bf16(a0, bU, uA[t][0]);
        gA[t][1] = wmma_bf16(a1, bG, gA[t][1]);
        uA[t][1] = wmma_bf16(a1, bU, uA[t][1]);
      }
    }
    // silu(gate)*up on f32 accumulators; pack bf16 into LDS tile.
    // C layout: element r -> M = rg*16 + r + 8*hi, N = lane&15.
#pragma unroll
    for (int t = 0; t < 2; ++t) {
      const int col = (ct0 + t) * 16 + mr;
#pragma unroll
      for (int rg = 0; rg < 2; ++rg) {
#pragma unroll
        for (int r = 0; r < 8; ++r) {
          float g = gA[t][rg][r];
          float a = (g / (1.0f + __expf(-g))) * uA[t][rg][r];
          s_a[rg * 16 + r + hi * 8][col] = f32_to_bf16(a);
        }
      }
    }
  }
  __syncthreads();

  // ---- phase 2: y = a @ w2e^T, streaming w2 once per workgroup ----
  const unsigned short* sRow0 = &s_a[mr][0];
  const unsigned short* sRow1 = &s_a[16 + mr][0];
  for (int p = 0; p < 8; ++p) {
    const int ot0 = (p * 8 + wave) * 2;          // out col tiles ot0, ot0+1
    const unsigned short* bw[2];
    bw[0] = w2e + (size_t)(ot0 * 16 + mr) * IDIM;
    bw[1] = bw[0] + (size_t)16 * IDIM;
    __builtin_prefetch(bw[0], 0, 0);
    __builtin_prefetch(bw[1], 0, 0);

    v8f acc[2][2];                               // [col-tile][row-group]
#pragma unroll
    for (int t = 0; t < 2; ++t)
#pragma unroll
      for (int rg = 0; rg < 2; ++rg) acc[t][rg] = v8f_zero();

    for (int k0 = 0; k0 < IDIM; k0 += 32) {
      v16bf a0 = load_a_frag(sRow0, k0, hi);     // ds_load_b128
      v16bf a1 = load_a_frag(sRow1, k0, hi);
#pragma unroll
      for (int t = 0; t < 2; ++t) {
        v16bf bF = load_b_frag(bw[t], k0, hi);
        acc[t][0] = wmma_bf16(a0, bF, acc[t][0]);
        acc[t][1] = wmma_bf16(a1, bF, acc[t][1]);
      }
    }
    // scale by router weight, combine (exactly 2 commutative adds/element)
#pragma unroll
    for (int t = 0; t < 2; ++t) {
      const int col = (ot0 + t) * 16 + mr;
#pragma unroll
      for (int rg = 0; rg < 2; ++rg) {
#pragma unroll
        for (int r = 0; r < 8; ++r) {
          const int Mrow = rg * 16 + r + hi * 8;
          atomicAdd(out + (size_t)s_token[Mrow] * KDIM + col,
                    acc[t][rg][r] * s_w[Mrow]);
        }
      }
    }
  }
}

extern "C" void kernel_launch(void* const* d_in, const int* in_sizes, int n_in,
                              void* d_out, int out_size, void* d_ws,
                              size_t ws_size, hipStream_t stream) {
  (void)in_sizes; (void)n_in; (void)out_size; (void)ws_size;
  const float* hs  = (const float*)d_in[0];  // [M,K]
  const float* w1  = (const float*)d_in[1];  // [E,2I,K]
  const float* w2  = (const float*)d_in[2];  // [E,K,I]
  const float* tw  = (const float*)d_in[3];  // [M,TOPK]
  const int*   ids = (const int*)d_in[4];    // [M,TOPK]
  float* out = (float*)d_out;
  char* ws = (char*)d_ws;

  // workspace layout
  int* counts = (int*)ws;                                    // E ints
  int* tok = (int*)(ws + 256);                               // E*MAXP ints
  float* wgt = (float*)(ws + 256 + sizeof(int) * (size_t)NEXP * MAXP);
  size_t offX = (256 + 2u * sizeof(int) * (size_t)NEXP * MAXP + 255) &
                ~(size_t)255;
  unsigned short* Xbf  = (unsigned short*)(ws + offX);
  unsigned short* W1bf = Xbf  + (size_t)MTOK * KDIM;
  unsigned short* W2bf = W1bf + (size_t)NEXP * 2 * IDIM * KDIM;

  const long long nX  = (long long)MTOK * KDIM;              // 8.4M
  const long long nW1 = (long long)NEXP * 2 * IDIM * KDIM;   // 67M
  const long long nW2 = (long long)NEXP * KDIM * IDIM;       // 33.5M

  zero_init<<<(MTOK * KDIM) / 4 / 256, 256, 0, stream>>>(out, counts);
  cvt_f32_to_bf16<<<(unsigned)(nX / 4 / 256), 256, 0, stream>>>(hs, Xbf, nX);
  cvt_f32_to_bf16<<<(unsigned)(nW1 / 4 / 256), 256, 0, stream>>>(w1, W1bf, nW1);
  cvt_f32_to_bf16<<<(unsigned)(nW2 / 4 / 256), 256, 0, stream>>>(w2, W2bf, nW2);
  build_lists<<<(MTOK * NTOPK) / 256, 256, 0, stream>>>(ids, tw, counts, tok,
                                                        wgt);
  dim3 grid(MAXTILES, NEXP);
  moe_main<<<grid, 256, 0, stream>>>(Xbf, W1bf, W2bf, counts, tok, wgt, out);
}
